// DeepSpeedMLP_31035433681240
// MI455X (gfx1250) — matrix operand order repaired and checked by reading, not compiled, not run on hardware
//
#include <hip/hip_runtime.h>
#include <math.h>

#define NTOK 4096   // B*S
#define HDIM 4096
#define IDIM 16384
#define LN_EPS 1e-5f

typedef __attribute__((ext_vector_type(16))) __bf16        v16bf;
typedef __attribute__((ext_vector_type(8)))  float         v8f;
typedef __attribute__((ext_vector_type(4)))  unsigned int  u32x4;
typedef __attribute__((ext_vector_type(8)))  int           i32x8;
typedef __attribute__((ext_vector_type(4)))  int           i32x4;

#if defined(__has_builtin)
#if __has_builtin(__builtin_amdgcn_tensor_load_to_lds) && \
    __has_builtin(__builtin_amdgcn_s_wait_tensorcnt)
#define USE_TDM 1
#endif
#endif
#ifndef USE_TDM
#define USE_TDM 0
#endif

__device__ __forceinline__ unsigned short f2bf(float x) {
    unsigned int u = __float_as_uint(x);
    unsigned int r = u + 0x7FFFu + ((u >> 16) & 1u);   // round-to-nearest-even
    return (unsigned short)(r >> 16);
}

#if USE_TDM
// TDM: copy a [64 rows x 32 bf16] tile (row stride = K elements) from global
// into LDS, inserting 16B of padding after every 64B row -> LDS stride 80B,
// matching the padded fragment addressing below. Descriptor per ISA 08 §8.
__device__ __forceinline__ void tdm_load_tile_bf16(unsigned lds_byte_off,
                                                   const unsigned short* gptr,
                                                   int K)
{
    unsigned long long ga = (unsigned long long)gptr;
    u32x4 g0;
    g0.x = 1u;                                    // count=1: valid user descriptor
    g0.y = lds_byte_off;                          // lds_addr
    g0.z = (unsigned)(ga & 0xFFFFFFFFull);        // global_addr[31:0]
    g0.w = (unsigned)((ga >> 32) & 0x01FFFFFFull) // global_addr[56:32]
         | (2u << 30);                            // type = 2 ("image")
    i32x8 g1;
    g1[0] = (int)((1u << 16)      // data_size = 2 bytes
                | (1u << 20)      // pad_enable
                | (3u << 22)      // pad_interval: 16 DWORDs (64B) between pads
                | (3u << 25));    // pad_amount: 4 DWORDs (16B)
    g1[1] = (int)(((unsigned)K & 0xFFFFu) << 16);            // tensor_dim0 lo16
    g1[2] = (int)((((unsigned)K >> 16) & 0xFFFFu)            // tensor_dim0 hi16
                | ((unsigned)(NTOK & 0xFFFF) << 16));        // tensor_dim1 lo16
    g1[3] = (int)(32u << 16);     // tensor_dim1 hi16 = 0 | tile_dim0 = 32
    g1[4] = 64;                   // tile_dim1 = 64 rows, tile_dim2 = 0
    g1[5] = K;                    // tensor_dim0_stride lo32 (elements)
    g1[6] = 0;                    // stride hi / dim1_stride (unused for 2D tile)
    g1[7] = 0;
    const i32x4 z4 = (i32x4)0;
    const i32x8 z8 = (i32x8)0;
    __builtin_amdgcn_tensor_load_to_lds(g0, g1, z4, z4, z8, 0);
}
#endif

// ---------------------------------------------------------------------------
// Kernel 1: pre = input + residual + bias ; ln = LN(pre)*nw + nb  -> bf16
// ---------------------------------------------------------------------------
__global__ void __launch_bounds__(256)
ln_kernel(const float* __restrict__ in, const float* __restrict__ res,
          const float* __restrict__ bias, const float* __restrict__ nw,
          const float* __restrict__ nb, unsigned short* __restrict__ ln_out)
{
    const int row = blockIdx.x;
    const int t = threadIdx.x;
    const float* pi = in  + (size_t)row * HDIM;
    const float* pr = res + (size_t)row * HDIM;

    float4 v[4];
    float s = 0.f, ss = 0.f;
#pragma unroll
    for (int c = 0; c < 4; ++c) {
        const int idx = c * 1024 + t * 4;
        float4 a = *(const float4*)(pi + idx);
        float4 b = *(const float4*)(pr + idx);
        float4 g = *(const float4*)(bias + idx);
        float4 p;
        p.x = a.x + b.x + g.x; p.y = a.y + b.y + g.y;
        p.z = a.z + b.z + g.z; p.w = a.w + b.w + g.w;
        v[c] = p;
        s  += p.x + p.y + p.z + p.w;
        ss += p.x*p.x + p.y*p.y + p.z*p.z + p.w*p.w;
    }

    __shared__ float red[512];
    red[t] = s; red[256 + t] = ss;
    __syncthreads();
#pragma unroll
    for (int o = 128; o > 0; o >>= 1) {
        if (t < o) { red[t] += red[t + o]; red[256 + t] += red[256 + t + o]; }
        __syncthreads();
    }
    const float mu   = red[0] * (1.0f / HDIM);
    const float var  = red[256] * (1.0f / HDIM) - mu * mu;
    const float rstd = rsqrtf(var + LN_EPS);

    unsigned short* po = ln_out + (size_t)row * HDIM;
#pragma unroll
    for (int c = 0; c < 4; ++c) {
        const int idx = c * 1024 + t * 4;
        float4 w4 = *(const float4*)(nw + idx);
        float4 b4 = *(const float4*)(nb + idx);
        ushort4 o;
        o.x = f2bf((v[c].x - mu) * rstd * w4.x + b4.x);
        o.y = f2bf((v[c].y - mu) * rstd * w4.y + b4.y);
        o.z = f2bf((v[c].z - mu) * rstd * w4.z + b4.z);
        o.w = f2bf((v[c].w - mu) * rstd * w4.w + b4.w);
        *(ushort4*)(po + idx) = o;
    }
}

// ---------------------------------------------------------------------------
// GEMM: C[m,n] = sum_k A[m,k]*W[k,n]   A: bf16 [M,K], W: fp32 [K,ldn]
// EPI==0: out_bf16 = gelu(C + colbias)                       (fc1)
// EPI==1: out_f32  = C + colbias + input + residual + abias  (fc2 + resid add)
// Block 64x128, BK=32, 8 waves, wave tile 32x32 = 2x2 WMMA, double-buffered
// LDS; A tile staged by TDM (async), B tile transposed+cvt through VGPRs.
// ---------------------------------------------------------------------------
template <int EPI>
__global__ void __launch_bounds__(256)
gemm_kernel(const unsigned short* __restrict__ A, const float* __restrict__ W,
            const int K, const int ldn,
            const float* __restrict__ colbias,
            const float* __restrict__ in0, const float* __restrict__ in1,
            const float* __restrict__ in2,
            unsigned short* __restrict__ outb, float* __restrict__ outf)
{
    constexpr int BM = 64, BN = 128, BK = 32, LDT = 40;  // LDT ushorts = 80B
    __shared__ __align__(16) unsigned short At[2][BM * LDT];
    __shared__ __align__(16) unsigned short Bt[2][BN * LDT];

    const int t    = threadIdx.x;
    const int m0   = blockIdx.y * BM;
    const int n0   = blockIdx.x * BN;
    const int wv   = t >> 5;
    const int lane = t & 31;
    const int wm   = (wv & 1) * 32;
    const int wn   = (wv >> 1) * 32;
    const int lrow = lane & 15;
    const int lhi  = lane >> 4;
    const int ar   = t >> 2;             // fallback A-stage row
    const int ac   = (t & 3) * 8;        // fallback A-stage col

    v8f acc[2][2] = {};

    auto stageA = [&](int buf, int k0) {
#if USE_TDM
        if (t < 32)   // one TDM issue per block (wave 0; EXEC ignored by TDM)
            tdm_load_tile_bf16((unsigned)(size_t)&At[buf][0],
                               A + (size_t)m0 * K + k0, K);
#else
        const uint4 src = *(const uint4*)(A + (size_t)(m0 + ar) * K + (k0 + ac));
        *(uint4*)(&At[buf][ar * LDT + ac]) = src;
#endif
    };
    auto stageB = [&](int buf, int k0) {
#pragma unroll
        for (int it = 0; it < 4; ++it) {
            const int lin = it * 256 + t;
            const int k   = lin >> 5;          // 0..31
            const int n   = (lin & 31) * 4;    // 0..124
            float4 w4 = *(const float4*)(W + (size_t)(k0 + k) * ldn + (n0 + n));
            unsigned short* bp = &Bt[buf][0];
            bp[(n + 0) * LDT + k] = f2bf(w4.x);
            bp[(n + 1) * LDT + k] = f2bf(w4.y);
            bp[(n + 2) * LDT + k] = f2bf(w4.z);
            bp[(n + 3) * LDT + k] = f2bf(w4.w);
            if (it == 0 && k0 + BK < K)
                __builtin_prefetch(W + (size_t)(k0 + BK + k) * ldn + (n0 + n), 0, 1);
        }
    };

    stageA(0, 0);
    stageB(0, 0);

    const int NIT = K / BK;
    for (int i = 0; i < NIT; ++i) {
        const int buf = i & 1;
#if USE_TDM
        __builtin_amdgcn_s_wait_tensorcnt(0);  // TDM of `buf` complete (wave0; NOP elsewhere)
#endif
        __syncthreads();                       // staged tile visible block-wide
        if (i + 1 < NIT) {                     // kick off next slab (overlaps WMMA)
            stageA(buf ^ 1, (i + 1) * BK);
            stageB(buf ^ 1, (i + 1) * BK);
        }

        union FragU { uint4 q[2]; v16bf v; };
        FragU a[2], b[2];
#pragma unroll
        for (int mi = 0; mi < 2; ++mi) {
            // A 16x32 layout: lanes 0-15 -> K runs [0..7],[16..23]; lanes 16-31 -> +8
            const unsigned short* p = &At[buf][(wm + mi * 16 + lrow) * LDT + lhi * 8];
            a[mi].q[0] = *(const uint4*)(p);
            a[mi].q[1] = *(const uint4*)(p + 16);
        }
#pragma unroll
        for (int ni = 0; ni < 2; ++ni) {
            // B 32x16 layout: lanes 0-15 -> K runs [0..7],[16..23]; lanes 16-31 -> +8
            const unsigned short* p = &Bt[buf][(wn + ni * 16 + lrow) * LDT + lhi * 8];
            b[ni].q[0] = *(const uint4*)(p);
            b[ni].q[1] = *(const uint4*)(p + 16);
        }
#pragma unroll
        for (int mi = 0; mi < 2; ++mi)
#pragma unroll
            for (int ni = 0; ni < 2; ++ni)
                acc[mi][ni] = __builtin_amdgcn_wmma_f32_16x16x32_bf16(
                    false, a[mi].v, false, b[ni].v,
                    (short)0, acc[mi][ni], false, false);
    }

    // epilogue: C/D layout -> VGPR r: M = r + (lane>=16)*8, N = lane&15
#pragma unroll
    for (int mi = 0; mi < 2; ++mi) {
#pragma unroll
        for (int ni = 0; ni < 2; ++ni) {
            const int col   = n0 + wn + ni * 16 + lrow;
            const int rbase = m0 + wm + mi * 16 + lhi * 8;
            const float cb  = colbias[col];
#pragma unroll
            for (int r = 0; r < 8; ++r) {
                const int row = rbase + r;
                float x = acc[mi][ni][r] + cb;
                if constexpr (EPI == 0) {
                    x = 0.5f * x * (1.0f + erff(x * 0.70710678118654752f));
                    outb[(size_t)row * ldn + col] = f2bf(x);
                } else {
                    const size_t off = (size_t)row * ldn + col;
                    outf[off] = x + in0[off] + in1[off] + in2[col];
                }
            }
        }
    }
}

// ---------------------------------------------------------------------------
extern "C" void kernel_launch(void* const* d_in, const int* in_sizes, int n_in,
                              void* d_out, int out_size, void* d_ws, size_t ws_size,
                              hipStream_t stream)
{
    const float* input    = (const float*)d_in[0];
    const float* residual = (const float*)d_in[1];
    const float* bias     = (const float*)d_in[3];
    const float* attn_nw  = (const float*)d_in[5];
    const float* attn_nb  = (const float*)d_in[6];
    const float* inter_w  = (const float*)d_in[7];
    const float* inter_b  = (const float*)d_in[8];
    const float* output_w = (const float*)d_in[9];
    const float* output_b = (const float*)d_in[10];

    unsigned short* ln = (unsigned short*)d_ws;                                    // 32 MB
    unsigned short* h  = (unsigned short*)((char*)d_ws + (size_t)NTOK * HDIM * 2); // 128 MB
    float* out = (float*)d_out;

    ln_kernel<<<NTOK, 256, 0, stream>>>(input, residual, bias, attn_nw, attn_nb, ln);

    gemm_kernel<0><<<dim3(IDIM / 128, NTOK / 64), 256, 0, stream>>>(
        ln, inter_w, HDIM, IDIM, inter_b,
        nullptr, nullptr, nullptr, h, nullptr);

    gemm_kernel<1><<<dim3(HDIM / 128, NTOK / 64), 256, 0, stream>>>(
        h, output_w, IDIM, HDIM, output_b,
        input, residual, bias, nullptr, out);
}
